// Network_37804302139667
// MI455X (gfx1250) — matrix operand order repaired and checked by reading, not compiled, not run on hardware
//
#include <hip/hip_runtime.h>
#include <hip/hip_bf16.h>

typedef __attribute__((ext_vector_type(16))) __bf16 bf16x16;
typedef __attribute__((ext_vector_type(8)))  __bf16 bf16x8;
typedef __attribute__((ext_vector_type(4)))  __bf16 bf16x4;
typedef __attribute__((ext_vector_type(8)))  float  f32x8;
typedef __attribute__((ext_vector_type(4)))  float  f32x4;

__device__ __forceinline__ f32x8 wmma_bf16(bf16x16 a, bf16x16 b, f32x8 c) {
    return __builtin_amdgcn_wmma_f32_16x16x32_bf16(false, a, false, b, (short)0, c, false, false);
}

__device__ __forceinline__ float sigmoidf_(float x) { return 1.0f / (1.0f + __expf(-x)); }

// Build A operand (16x32 bf16 tile) from an f32 row-major matrix.
// Lane L: M = L&15; K runs [half*8, +8) and [half*8+16, +8), half = L>>4.
__device__ __forceinline__ bf16x16 load_a_f32(const float* __restrict__ base, int ld,
                                              int rowBase, int kb, int lan16, int half) {
    const float* p0 = base + (size_t)(rowBase + lan16) * ld + kb + half * 8;
    f32x4 a0 = *(const f32x4*)(p0);
    f32x4 a1 = *(const f32x4*)(p0 + 4);
    f32x4 b0 = *(const f32x4*)(p0 + 16);
    f32x4 b1 = *(const f32x4*)(p0 + 20);
    bf16x16 r;
#pragma unroll
    for (int i = 0; i < 4; ++i) {
        r[i]      = (__bf16)a0[i];
        r[4 + i]  = (__bf16)a1[i];
        r[8 + i]  = (__bf16)b0[i];
        r[12 + i] = (__bf16)b1[i];
    }
    return r;
}

// A operand from a bf16 row-major matrix.
__device__ __forceinline__ bf16x16 load_a_bf16(const __bf16* __restrict__ base, int ld,
                                               int rowBase, int kb, int lan16, int half) {
    const __bf16* p0 = base + (size_t)(rowBase + lan16) * ld + kb + half * 8;
    bf16x8 lo = *(const bf16x8*)(p0);
    bf16x8 hi = *(const bf16x8*)(p0 + 16);
    bf16x16 r;
#pragma unroll
    for (int i = 0; i < 8; ++i) { r[i] = lo[i]; r[8 + i] = hi[i]; }
    return r;
}

// B operand (32x16): lane holds column N = L&15; K runs [half*8, +8) and [half*8+16, +8).
__device__ __forceinline__ bf16x16 load_b(const __bf16* __restrict__ W, int ld,
                                          int rowN, int kb, int half) {
    const __bf16* p0 = W + (size_t)rowN * ld + kb + half * 8;
    bf16x8 lo = *(const bf16x8*)(p0);
    bf16x8 hi = *(const bf16x8*)(p0 + 16);
    bf16x16 r;
#pragma unroll
    for (int i = 0; i < 8; ++i) { r[i] = lo[i]; r[8 + i] = hi[i]; }
    return r;
}

__global__ void cvt_f32_bf16(const float* __restrict__ s, __bf16* __restrict__ d, int n4) {
    int i = blockIdx.x * blockDim.x + threadIdx.x;
    if (i < n4) {
        f32x4 v = ((const f32x4*)s)[i];
        bf16x4 o;
#pragma unroll
        for (int k = 0; k < 4; ++k) o[k] = (__bf16)v[k];
        ((bf16x4*)d)[i] = o;
    }
}

// One fused LSTM layer. gates = [X,H] @ [Wih|Whh]^T + (bih+bhh); gate order i,f,g,o.
// Each wave: 32 batch rows x 16 hidden units; 8 accumulators (4 gates x 2 row tiles).
// Every B (weight) fragment is reused by two WMMAs -> halves weight traffic from L2.
__global__ __launch_bounds__(256) void lstm_layer_kernel(
    const float* __restrict__ X, int K1,          // B x K1
    const float* __restrict__ H,                  // B x 1024
    const __bf16* __restrict__ Wih,               // 4096 x K1 (bf16)
    const __bf16* __restrict__ Whh,               // 4096 x 1024 (bf16)
    const float* __restrict__ bih, const float* __restrict__ bhh,  // 4096
    const float* __restrict__ Cin,                // B x 1024
    float* __restrict__ Hout, float* __restrict__ Cout)            // B x 1024
{
    const int L = 1024;
    int lane = threadIdx.x, wave = threadIdx.y;
    int half = lane >> 4, lan16 = lane & 15;
    int gid = blockIdx.x * 8 + wave;
    int u0 = (gid & 63) * 16;          // 64 unit tiles
    int rowBase = (gid >> 6) * 32;     // 256 row tiles of 32 rows

    const f32x8 zero = {0.f, 0.f, 0.f, 0.f, 0.f, 0.f, 0.f, 0.f};
    f32x8 accA[4] = {zero, zero, zero, zero};   // rows [rowBase, rowBase+16)
    f32x8 accB[4] = {zero, zero, zero, zero};   // rows [rowBase+16, rowBase+32)

    for (int kk = 0; kk < K1; kk += 32) {
        bf16x16 a0 = load_a_f32(X, K1, rowBase,      kk, lan16, half);
        bf16x16 a1 = load_a_f32(X, K1, rowBase + 16, kk, lan16, half);
#pragma unroll
        for (int g = 0; g < 4; ++g) {
            bf16x16 b = load_b(Wih, K1, g * L + u0 + lan16, kk, half);
            accA[g] = wmma_bf16(a0, b, accA[g]);
            accB[g] = wmma_bf16(a1, b, accB[g]);
        }
    }
    for (int kk = 0; kk < L; kk += 32) {
        bf16x16 a0 = load_a_f32(H, L, rowBase,      kk, lan16, half);
        bf16x16 a1 = load_a_f32(H, L, rowBase + 16, kk, lan16, half);
#pragma unroll
        for (int g = 0; g < 4; ++g) {
            bf16x16 b = load_b(Whh, L, g * L + u0 + lan16, kk, half);
            accA[g] = wmma_bf16(a0, b, accA[g]);
            accB[g] = wmma_bf16(a1, b, accB[g]);
        }
    }

    int col = u0 + lan16;
    float biasI = bih[0 * L + col] + bhh[0 * L + col];
    float biasF = bih[1 * L + col] + bhh[1 * L + col];
    float biasG = bih[2 * L + col] + bhh[2 * L + col];
    float biasO = bih[3 * L + col] + bhh[3 * L + col];
#pragma unroll
    for (int rs = 0; rs < 2; ++rs) {
        const f32x8* acc = rs ? accB : accA;
        int rb = rowBase + rs * 16;
#pragma unroll
        for (int j = 0; j < 8; ++j) {
            int row = rb + j + 8 * half;
            size_t idx = (size_t)row * L + col;
            float cold = Cin[idx];
            float iv = sigmoidf_(acc[0][j] + biasI);
            float fv = sigmoidf_(acc[1][j] + biasF);
            float gv = tanhf(acc[2][j] + biasG);
            float ov = sigmoidf_(acc[3][j] + biasO);
            float cn = fv * cold + iv * gv;
            Cout[idx] = cn;
            Hout[idx] = ov * tanhf(cn);
        }
    }
}

// mid = x @ Wd^T + bd + y3, stored bf16 into workspace. Wave: 16 rows x 64 cols.
__global__ __launch_bounds__(256) void mid_kernel(
    const float* __restrict__ X,            // 8192 x 512
    const __bf16* __restrict__ Wdb,         // 1024 x 512
    const float* __restrict__ bd,           // 1024
    const float* __restrict__ Y3,           // 8192 x 1024
    __bf16* __restrict__ midbf)             // 8192 x 1024
{
    int lane = threadIdx.x, wave = threadIdx.y;
    int half = lane >> 4, lan16 = lane & 15;
    int gid = blockIdx.x * 8 + wave;
    int us = (gid & 15) * 64;          // 16 strips of 64 cols
    int rowBase = (gid >> 4) * 16;
    const f32x8 zero = {0.f, 0.f, 0.f, 0.f, 0.f, 0.f, 0.f, 0.f};
    f32x8 acc[4] = {zero, zero, zero, zero};
    for (int kk = 0; kk < 512; kk += 32) {
        bf16x16 a = load_a_f32(X, 512, rowBase, kk, lan16, half);
#pragma unroll
        for (int t = 0; t < 4; ++t) {
            bf16x16 b = load_b(Wdb, 512, us + t * 16 + lan16, kk, half);
            acc[t] = wmma_bf16(a, b, acc[t]);
        }
    }
#pragma unroll
    for (int t = 0; t < 4; ++t) {
        int colN = us + t * 16 + lan16;
        float bb = bd[colN];
#pragma unroll
        for (int j = 0; j < 8; ++j) {
            int row = rowBase + j + 8 * half;
            size_t idx = (size_t)row * 1024 + colN;
            midbf[idx] = (__bf16)(acc[t][j] + bb + Y3[idx]);
        }
    }
}

// motors = mid @ Wm^T + bm, fused log_softmax over the 512 cols.
// Block handles 16 rows x all 512 cols (8 waves x 4 tiles).
__global__ __launch_bounds__(256) void motors_kernel(
    const __bf16* __restrict__ midbf,       // 8192 x 1024
    const __bf16* __restrict__ Wmb,         // 512 x 1024
    const float* __restrict__ bm,           // 512
    float* __restrict__ out)                // 8192 x 512
{
    __shared__ float sm[16][512];
    __shared__ float red[16][16];
    __shared__ float rowstat[16];
    int lane = threadIdx.x, wave = threadIdx.y;
    int half = lane >> 4, lan16 = lane & 15;
    int rowBase = blockIdx.x * 16;
    int us = wave * 64;
    const f32x8 zero = {0.f, 0.f, 0.f, 0.f, 0.f, 0.f, 0.f, 0.f};
    f32x8 acc[4] = {zero, zero, zero, zero};
    for (int kk = 0; kk < 1024; kk += 32) {
        bf16x16 a = load_a_bf16(midbf, 1024, rowBase, kk, lan16, half);
#pragma unroll
        for (int t = 0; t < 4; ++t) {
            bf16x16 b = load_b(Wmb, 1024, us + t * 16 + lan16, kk, half);
            acc[t] = wmma_bf16(a, b, acc[t]);
        }
    }
#pragma unroll
    for (int t = 0; t < 4; ++t) {
        int colN = us + t * 16 + lan16;
        float bb = bm[colN];
#pragma unroll
        for (int j = 0; j < 8; ++j) sm[j + 8 * half][colN] = acc[t][j] + bb;
    }
    __syncthreads();
    int tid = wave * 32 + lane;
    int r = tid >> 4, c0 = tid & 15;
    float m = -3.0e38f;
    for (int c = c0; c < 512; c += 16) m = fmaxf(m, sm[r][c]);
    red[r][c0] = m;
    __syncthreads();
    if (c0 == 0) {
        float mm = red[r][0];
        for (int i = 1; i < 16; ++i) mm = fmaxf(mm, red[r][i]);
        rowstat[r] = mm;
    }
    __syncthreads();
    float mm = rowstat[r];
    float s = 0.f;
    for (int c = c0; c < 512; c += 16) s += __expf(sm[r][c] - mm);
    red[r][c0] = s;
    __syncthreads();
    if (c0 == 0) {
        float ss = 0.f;
        for (int i = 0; i < 16; ++i) ss += red[r][i];
        rowstat[r] = mm + logf(ss);
    }
    __syncthreads();
    for (int idx = tid; idx < 16 * 512; idx += 256) {
        int rr = idx >> 9, cc = idx & 511;
        out[(size_t)(rowBase + rr) * 512 + cc] = sm[rr][cc] - rowstat[rr];
    }
}

extern "C" void kernel_launch(void* const* d_in, const int* in_sizes, int n_in,
                              void* d_out, int out_size, void* d_ws, size_t ws_size,
                              hipStream_t stream) {
    (void)in_sizes; (void)n_in; (void)out_size; (void)ws_size;
    const float* x    = (const float*)d_in[0];
    const float* h1   = (const float*)d_in[1];
    const float* c1   = (const float*)d_in[2];
    const float* h2   = (const float*)d_in[3];
    const float* c2   = (const float*)d_in[4];
    const float* h3   = (const float*)d_in[5];
    const float* c3   = (const float*)d_in[6];
    const float* Wih1 = (const float*)d_in[7];
    const float* Whh1 = (const float*)d_in[8];
    const float* bih1 = (const float*)d_in[9];
    const float* bhh1 = (const float*)d_in[10];
    const float* Wih2 = (const float*)d_in[11];
    const float* Whh2 = (const float*)d_in[12];
    const float* bih2 = (const float*)d_in[13];
    const float* bhh2 = (const float*)d_in[14];
    const float* Wih3 = (const float*)d_in[15];
    const float* Whh3 = (const float*)d_in[16];
    const float* bih3 = (const float*)d_in[17];
    const float* bhh3 = (const float*)d_in[18];
    const float* Wd   = (const float*)d_in[19];
    const float* bd   = (const float*)d_in[20];
    const float* Wm   = (const float*)d_in[21];
    const float* bm   = (const float*)d_in[22];

    // Workspace layout (bf16 weights + bf16 mid), offsets in bytes.
    char* ws = (char*)d_ws;
    __bf16* Wih1b = (__bf16*)(ws + 0);           // 4096x512
    __bf16* Whh1b = (__bf16*)(ws + 4194304);     // 4096x1024
    __bf16* Wih2b = (__bf16*)(ws + 12582912);    // 4096x1024
    __bf16* Whh2b = (__bf16*)(ws + 20971520);    // 4096x1024
    __bf16* Wih3b = (__bf16*)(ws + 29360128);    // 4096x1024
    __bf16* Whh3b = (__bf16*)(ws + 37748736);    // 4096x1024
    __bf16* Wdb   = (__bf16*)(ws + 46137344);    // 1024x512
    __bf16* Wmb   = (__bf16*)(ws + 47185920);    // 512x1024
    __bf16* midbf = (__bf16*)(ws + 48234496);    // 8192x1024

    auto cvt = [&](const float* s, __bf16* d, int n) {
        int n4 = n / 4;
        cvt_f32_bf16<<<(n4 + 255) / 256, 256, 0, stream>>>(s, d, n4);
    };
    cvt(Wih1, Wih1b, 4096 * 512);
    cvt(Whh1, Whh1b, 4096 * 1024);
    cvt(Wih2, Wih2b, 4096 * 1024);
    cvt(Whh2, Whh2b, 4096 * 1024);
    cvt(Wih3, Wih3b, 4096 * 1024);
    cvt(Whh3, Whh3b, 4096 * 1024);
    cvt(Wd,   Wdb,   1024 * 512);
    cvt(Wm,   Wmb,   512 * 1024);

    // Output slices (floats): motor_preds, y1, nc1, y2, nc2, y3, nc3.
    float* out = (float*)d_out;
    float* y1  = out + 4194304;
    float* nc1 = y1 + 8388608;
    float* y2  = nc1 + 8388608;
    float* nc2 = y2 + 8388608;
    float* y3  = nc2 + 8388608;
    float* nc3 = y3 + 8388608;

    dim3 blk(32, 8);
    // 256 row tiles (32 rows) x 64 unit tiles = 16384 waves = 2048 blocks.
    lstm_layer_kernel<<<2048, blk, 0, stream>>>(x,  512,  h1, Wih1b, Whh1b, bih1, bhh1, c1, y1, nc1);
    lstm_layer_kernel<<<2048, blk, 0, stream>>>(y1, 1024, h2, Wih2b, Whh2b, bih2, bhh2, c2, y2, nc2);
    lstm_layer_kernel<<<2048, blk, 0, stream>>>(y2, 1024, h3, Wih3b, Whh3b, bih3, bhh3, c3, y3, nc3);
    mid_kernel<<<1024, blk, 0, stream>>>(x, Wdb, bd, y3, midbf);
    motors_kernel<<<512, blk, 0, stream>>>(midbf, Wmb, bm, out);
}
